// SubmatrixLinear_81956565942410
// MI455X (gfx1250) — compile-verified, matrix operand order checked by reading, not run on hardware
//
#include <hip/hip_runtime.h>
#include <hip/hip_bf16.h>

typedef __bf16 bf16_t;
typedef __attribute__((ext_vector_type(16))) bf16_t v16bf;
typedef __attribute__((ext_vector_type(8)))  float  v8f;
typedef unsigned short ushort_t;
typedef unsigned int   uint_t;
typedef int v4i_t __attribute__((__vector_size__(16)));

// ---------------- bf16 split helpers (RNE) ----------------
__device__ __forceinline__ ushort_t f2bf(float f) {
  uint_t u = __float_as_uint(f);
  u += 0x7FFFu + ((u >> 16) & 1u);
  return (ushort_t)(u >> 16);
}
__device__ __forceinline__ float bf2f(ushort_t h) {
  return __uint_as_float(((uint_t)h) << 16);
}

union FragU { uint4 q[2]; v16bf v; };

// ---------------- CDNA5 async global->LDS copy (ASYNCcnt path) ----------------
#if __has_builtin(__builtin_amdgcn_global_load_async_to_lds_b128)
#define HAVE_ASYNC_LDS 1
#define ASYNC_CP128(gsrc, ldst)                                          \
  __builtin_amdgcn_global_load_async_to_lds_b128(                        \
      (__attribute__((address_space(1))) v4i_t*)(gsrc),                  \
      (__attribute__((address_space(3))) v4i_t*)(ldst), 0, 0)
#else
#define HAVE_ASYNC_LDS 0
#endif

#define WAIT_ASYNC() asm volatile("s_wait_asynccnt 0x0" ::: "memory")
#define WAIT_DS()    asm volatile("s_wait_dscnt 0x0" ::: "memory")

// ---------------- workspace layout (bytes) ----------------
#define WS_GATES   (0u)                       // 8192*16 f32      = 512KB
#define WS_COEF_HI (524288u)                  // 8192*128 bf16    = 2MB
#define WS_COEF_LO (2621440u)                 // 2MB
#define WS_XHI     (4718592u)                 // 8192*2048 bf16   = 32MB
#define WS_XLO     (38273024u)                // 32MB
#define WS_WHI     (71827456u)                // 2048*2176 bf16   = 8.5MB
#define WS_WLO     (80740352u)                // 8.5MB
#define WS_BPHI    (89653248u)                // 128*2048 bf16    = 512KB
#define WS_BPLO    (90177536u)                // 512KB  (end ~86.5MB)

// ---------------- f32 -> (bf16 hi, bf16 lo) splitter ----------------
__global__ __launch_bounds__(256) void split_f32_kernel(
    const float* __restrict__ src, ushort_t* __restrict__ hi,
    ushort_t* __restrict__ lo, int n4) {
  int i = blockIdx.x * 256 + threadIdx.x;
  if (i >= n4) return;
  float4 v = ((const float4*)src)[i];
  ushort4 hv, lv;
  hv.x = f2bf(v.x); lv.x = f2bf(v.x - bf2f(hv.x));
  hv.y = f2bf(v.y); lv.y = f2bf(v.y - bf2f(hv.y));
  hv.z = f2bf(v.z); lv.z = f2bf(v.z - bf2f(hv.z));
  hv.w = f2bf(v.w); lv.w = f2bf(v.w - bf2f(hv.w));
  ((ushort4*)hi)[i] = hv;
  ((ushort4*)lo)[i] = lv;
}

// ---------------- pack Wcat = [Wb | A_flat] as hi/lo bf16, [2048][2176] ----------------
__global__ __launch_bounds__(256) void pack_wcat_kernel(
    const float* __restrict__ Wb, const float* __restrict__ A,
    ushort_t* __restrict__ whi, ushort_t* __restrict__ wlo) {
  int c = blockIdx.x * 256 + threadIdx.x;   // 0..2175 (K index)
  int o = blockIdx.y;                       // 0..2047 (output col)
  if (c >= 2176) return;
  float v;
  if (c < 2048) {
    v = Wb[(size_t)o * 2048 + c];
  } else {
    int kr = c - 2048;
    int k = kr >> 3, rr = kr & 7;           // A[k][o][r]
    v = A[((size_t)k * 2048 + o) * 8 + rr];
  }
  ushort_t h = f2bf(v);
  whi[(size_t)o * 2176 + c] = h;
  wlo[(size_t)o * 2176 + c] = f2bf(v - bf2f(h));
}

// ---------------- gating kernel: q = xWq^T + bq ; softmax+threshold+renorm ----------------
__global__ __launch_bounds__(128) void gate_kernel(
    const float* __restrict__ x, const float* __restrict__ Wq,
    const float* __restrict__ bq, const float* __restrict__ keys,
    float* __restrict__ gates) {
  __shared__ float qsh[4][32];
  int wave = threadIdx.x >> 5, lane = threadIdx.x & 31;
  int n = blockIdx.x * 4 + wave;
  const float* xr = x + (size_t)n * 2048;
  const float* wr = Wq + (size_t)lane * 2048;  // lane == kd
  float acc = 0.f;
  for (int i0 = 0; i0 < 2048; i0 += 32) {
    float xv = xr[i0 + lane];
    #pragma unroll
    for (int j = 0; j < 32; ++j) {
      float xj = __shfl(xv, j, 32);
      acc += xj * wr[i0 + j];
    }
  }
  acc += bq[lane];
  qsh[wave][lane] = acc;
  __syncthreads();
  float s = -1e30f;
  if (lane < 16) {
    const float* kr = keys + lane * 32;
    float d = 0.f;
    #pragma unroll
    for (int j = 0; j < 32; ++j) d += qsh[wave][j] * kr[j];
    s = d * 0.17677669529663687f;  // 1/sqrt(32)
  }
  // softmax over lanes 0..15 (xor stays inside 16-lane half)
  float m = s;
  #pragma unroll
  for (int d = 1; d < 16; d <<= 1) m = fmaxf(m, __shfl_xor(m, d, 32));
  float e = __expf(s - m);
  float sum = e;
  #pragma unroll
  for (int d = 1; d < 16; d <<= 1) sum += __shfl_xor(sum, d, 32);
  float g = e / sum;
  g = (g > 0.05f) ? g : 0.f;
  float s2 = g;
  #pragma unroll
  for (int d = 1; d < 16; d <<= 1) s2 += __shfl_xor(s2, d, 32);
  g = g / fmaxf(s2, 1e-8f);
  if (lane < 16) gates[n * 16 + lane] = g;
}

// ---------------- bf16x3 WMMA GEMM, async double-buffered staging ----------------
// Block: 256 threads = 8 wave32, tile 128(M) x 128(N), K step = 32.
// A source: rows [M][K] bf16 hi/lo (x, or coef for K >= k_split).
// B source: rows [N][K] bf16 hi/lo (Wcat / Bpack, row-major N-by-K).
// MODE 0: out[m][n] = acc + bias[n]          (main GEMM, out_stride cols)
// MODE 1: coef[m][n] = acc * gates[m][n>>3]  -> bf16 hi/lo
template<int MODE>
__global__ __launch_bounds__(256) void gemm_bf16x3_kernel(
    const ushort_t* __restrict__ a_hi, const ushort_t* __restrict__ a_lo, int a_stride,
    const ushort_t* __restrict__ a2_hi, const ushort_t* __restrict__ a2_lo, int a2_stride,
    int k_split, int k_total,
    const ushort_t* __restrict__ b_hi, const ushort_t* __restrict__ b_lo, int b_stride,
    const float* __restrict__ bias, const float* __restrict__ gates,
    float* __restrict__ out, int out_stride,
    ushort_t* __restrict__ c_hi, ushort_t* __restrict__ c_lo) {
  // [buf][A/B][hi/lo][subtile 16rows][16x32] = 64KB
  __shared__ ushort_t lds[2][2][2][8][512];
  const int tid   = threadIdx.x;
  const int lane  = tid & 31;
  const int wave  = tid >> 5;
  const int waveM = wave >> 2;  // 0..1
  const int waveN = wave & 3;   // 0..3
  const int r     = lane & 15;
  const int h     = lane >> 4;
  const int blockM = blockIdx.y;
  const int blockN = blockIdx.x;

  v8f acc[4][2];
  #pragma unroll
  for (int m = 0; m < 4; ++m)
    #pragma unroll
    for (int n = 0; n < 2; ++n)
      #pragma unroll
      for (int e = 0; e < 8; ++e) acc[m][n][e] = 0.f;

  // Issue staging for K-offset kk into LDS buffer b.
  // Each thread moves 2x16B (A hi+lo) + 2x16B (B hi+lo) x 2 chunks = 8 b128 copies.
  auto stage = [&](int kk, int b) {
    const ushort_t* shi; const ushort_t* slo; int str; int col;
    if (kk < k_split) { shi = a_hi;  slo = a_lo;  str = a_stride;  col = kk; }
    else              { shi = a2_hi; slo = a2_lo; str = a2_stride; col = kk - k_split; }
    #pragma unroll
    for (int cc = 0; cc < 2; ++cc) {
      int c   = tid + cc * 256;    // chunk of 8 ushorts, 0..511
      int row = c >> 2;            // 0..127
      int c8  = (c & 3) * 8;       // 0,8,16,24
      int loff = (row & 15) * 32 + c8;
      int st   = row >> 4;
      {
        size_t goff = (size_t)(blockM * 128 + row) * str + col + c8;
#if HAVE_ASYNC_LDS
        ASYNC_CP128(shi + goff, &lds[b][0][0][st][loff]);
        ASYNC_CP128(slo + goff, &lds[b][0][1][st][loff]);
#else
        *(uint4*)&lds[b][0][0][st][loff] = *(const uint4*)(shi + goff);
        *(uint4*)&lds[b][0][1][st][loff] = *(const uint4*)(slo + goff);
#endif
      }
      {
        size_t goff = (size_t)(blockN * 128 + row) * b_stride + kk + c8;
#if HAVE_ASYNC_LDS
        ASYNC_CP128(b_hi + goff, &lds[b][1][0][st][loff]);
        ASYNC_CP128(b_lo + goff, &lds[b][1][1][st][loff]);
#else
        *(uint4*)&lds[b][1][0][st][loff] = *(const uint4*)(b_hi + goff);
        *(uint4*)&lds[b][1][1][st][loff] = *(const uint4*)(b_lo + goff);
#endif
      }
    }
  };

  stage(0, 0);
  int buf = 0;

  #pragma unroll 1
  for (int kk = 0; kk < k_total; kk += 32) {
    WAIT_ASYNC();           // my async copies into lds[buf] have landed
    __syncthreads();        // ...and everyone else's too
    if (kk + 32 < k_total) stage(kk + 32, buf ^ 1);  // overlap next fetch with compute

    // ---- load fragments (ISA 16-bit 16x32 layout: half-lane K split) ----
    FragU a[4][2], b[2][2];
    #pragma unroll
    for (int m = 0; m < 4; ++m)
      #pragma unroll
      for (int hl = 0; hl < 2; ++hl) {
        const uint_t* t = (const uint_t*)&lds[buf][0][hl][waveM * 4 + m][0];
        a[m][hl].q[0] = *(const uint4*)(t + r * 16 + h * 4);
        a[m][hl].q[1] = *(const uint4*)(t + r * 16 + h * 4 + 8);
      }
    #pragma unroll
    for (int n = 0; n < 2; ++n)
      #pragma unroll
      for (int hl = 0; hl < 2; ++hl) {
        const uint_t* t = (const uint_t*)&lds[buf][1][hl][waveN * 2 + n][0];
        b[n][hl].q[0] = *(const uint4*)(t + r * 16 + h * 4);
        b[n][hl].q[1] = *(const uint4*)(t + r * 16 + h * 4 + 8);
      }
    // ---- 24 WMMAs: hi*hi + hi*lo + lo*hi ----
    #pragma unroll
    for (int m = 0; m < 4; ++m)
      #pragma unroll
      for (int n = 0; n < 2; ++n) {
        acc[m][n] = __builtin_amdgcn_wmma_f32_16x16x32_bf16(
            false, a[m][0].v, false, b[n][0].v, (short)0, acc[m][n], false, false);
        acc[m][n] = __builtin_amdgcn_wmma_f32_16x16x32_bf16(
            false, a[m][0].v, false, b[n][1].v, (short)0, acc[m][n], false, false);
        acc[m][n] = __builtin_amdgcn_wmma_f32_16x16x32_bf16(
            false, a[m][1].v, false, b[n][0].v, (short)0, acc[m][n], false, false);
      }

    WAIT_DS();              // my frag reads of lds[buf] are done
    __syncthreads();        // everyone done reading -> buf may be overwritten next iter
    buf ^= 1;
  }

  // ---- epilogue (C/D layout: vgpr e -> M = base + e (+8 for lanes 16-31), N = lane%16)
  #pragma unroll
  for (int m = 0; m < 4; ++m) {
    int mbase = blockM * 128 + (waveM * 4 + m) * 16 + h * 8;
    #pragma unroll
    for (int n = 0; n < 2; ++n) {
      int nn = blockN * 128 + (waveN * 2 + n) * 16 + r;
      if (MODE == 0) {
        float bv = bias[nn];
        #pragma unroll
        for (int e = 0; e < 8; ++e)
          out[(size_t)(mbase + e) * out_stride + nn] = acc[m][n][e] + bv;
      } else {
        int kidx = nn >> 3;
        #pragma unroll
        for (int e = 0; e < 8; ++e) {
          int mm = mbase + e;
          float g = gates[mm * 16 + kidx];
          float val = acc[m][n][e] * g;
          ushort_t hh = f2bf(val);
          c_hi[(size_t)mm * 128 + nn] = hh;
          c_lo[(size_t)mm * 128 + nn] = f2bf(val - bf2f(hh));
        }
      }
    }
  }
}

extern "C" void kernel_launch(void* const* d_in, const int* in_sizes, int n_in,
                              void* d_out, int out_size, void* d_ws, size_t ws_size,
                              hipStream_t stream) {
  (void)in_sizes; (void)n_in; (void)out_size; (void)ws_size;
  const float* x    = (const float*)d_in[0];
  const float* Wb   = (const float*)d_in[1];
  const float* bb   = (const float*)d_in[2];
  const float* Wq   = (const float*)d_in[3];
  const float* bq   = (const float*)d_in[4];
  const float* A    = (const float*)d_in[5];
  const float* Bm   = (const float*)d_in[6];
  const float* keys = (const float*)d_in[7];
  float* out = (float*)d_out;
  char* ws = (char*)d_ws;

  float*    gates  = (float*)(ws + WS_GATES);
  ushort_t* coefhi = (ushort_t*)(ws + WS_COEF_HI);
  ushort_t* coeflo = (ushort_t*)(ws + WS_COEF_LO);
  ushort_t* xhi    = (ushort_t*)(ws + WS_XHI);
  ushort_t* xlo    = (ushort_t*)(ws + WS_XLO);
  ushort_t* whi    = (ushort_t*)(ws + WS_WHI);
  ushort_t* wlo    = (ushort_t*)(ws + WS_WLO);
  ushort_t* bphi   = (ushort_t*)(ws + WS_BPHI);
  ushort_t* bplo   = (ushort_t*)(ws + WS_BPLO);

  // 1) split x and B (B is [K][R][IN] = flat [128][2048]) into bf16 hi/lo
  split_f32_kernel<<<16384, 256, 0, stream>>>(x,  xhi,  xlo,  (8192 * 2048) / 4);
  split_f32_kernel<<<256,   256, 0, stream>>>(Bm, bphi, bplo, (128 * 2048) / 4);
  // 2) pack Wcat = [Wb | A_flat] hi/lo, [2048][2176]
  pack_wcat_kernel<<<dim3(9, 2048), 256, 0, stream>>>(Wb, A, whi, wlo);
  // 3) gates
  gate_kernel<<<2048, 128, 0, stream>>>(x, Wq, bq, keys, gates);
  // 4) coef = gate * (x @ B^T): M=8192, N=128, K=2048
  gemm_bf16x3_kernel<1><<<dim3(1, 64), 256, 0, stream>>>(
      xhi, xlo, 2048, (const ushort_t*)nullptr, (const ushort_t*)nullptr, 0,
      2048, 2048, bphi, bplo, 2048,
      (const float*)nullptr, gates, (float*)nullptr, 0, coefhi, coeflo);
  // 5) out = [x | coef] @ [Wb^T ; A_flat] + bb: M=8192, N=2048, K=2176
  gemm_bf16x3_kernel<0><<<dim3(16, 64), 256, 0, stream>>>(
      xhi, xlo, 2048, coefhi, coeflo, 128,
      2048, 2176, whi, wlo, 2176,
      bb, (const float*)nullptr, out, 2048, (ushort_t*)nullptr, (ushort_t*)nullptr);
}
